// HGTModule_27616639713819
// MI455X (gfx1250) — matrix operand order repaired and checked by reading, not compiled, not run on hardware
//
#include <hip/hip_runtime.h>
#include <math.h>

// ---- problem constants (match reference) ----
#define HID 128
#define NHEAD 8
#define DH 16
#define NLAYER 2
#define NP 100000
#define NA 50000
#define NEDGE 300000
#define IN_P 256
#define IN_A 128

typedef __attribute__((ext_vector_type(16))) __bf16 v16bf;
typedef __attribute__((ext_vector_type(8)))  __bf16 v8bf;
typedef __attribute__((ext_vector_type(8)))  float  v8f;

// ---------------- helpers ----------------
__device__ __forceinline__ __bf16 f2bf(float f) {
    union { float f; unsigned u; } v; v.f = f;
    unsigned r = v.u + 0x7FFFu + ((v.u >> 16) & 1u);   // round-to-nearest-even
    unsigned short h = (unsigned short)(r >> 16);
    __bf16 o; __builtin_memcpy(&o, &h, 2); return o;
}
// monotonic float<->uint encoding for atomicMax-based segment max
__device__ __forceinline__ unsigned encf(float f) {
    unsigned u = __float_as_uint(f);
    return (u & 0x80000000u) ? ~u : (u | 0x80000000u);
}
__device__ __forceinline__ float decf(unsigned e) {
    return (e & 0x80000000u) ? __uint_as_float(e & 0x7FFFFFFFu) : __uint_as_float(~e);
}

// ---------------- fills / converts ----------------
__global__ void fill_f32(float* p, float v, int n) {
    int i = blockIdx.x * blockDim.x + threadIdx.x;
    if (i < n) p[i] = v;
}
__global__ void fill_u32(unsigned* p, unsigned v, int n) {
    int i = blockIdx.x * blockDim.x + threadIdx.x;
    if (i < n) p[i] = v;
}
// Convert a batch of [K,128] f32 matrices into [128,K] bf16 (transposed, fragment-friendly).
__global__ void cvt_transpose_bf16(const float* __restrict__ s, __bf16* __restrict__ d,
                                   int K, int total) {
    int i = blockIdx.x * blockDim.x + threadIdx.x;
    if (i >= total) return;
    int msz = K * 128;
    int m = i / msz, rem = i - m * msz;
    int k = rem >> 7, n = rem & 127;
    d[(size_t)m * msz + (size_t)n * K + k] = f2bf(s[i]);
}

// ---------------- WMMA GEMM: out[M,128] = act(A[M,K] @ W[K,128] + bias) ----------------
// Wt is the K x 128 weight stored TRANSPOSED as [128][K] bf16 so each lane's B
// fragment (16 consecutive k at fixed n) is two contiguous b128 loads.
// mode 0: none   1: relu   2: skip-blend  a*val + (1-a)*xold, a=sigmoid(*skip)
// One block = 16 rows x 128 cols; 8 waves, each owns a 16x16 WMMA tile along N.
template <int K>
__global__ __launch_bounds__(256) void gemm_bf16_n128(
    const float* __restrict__ A, const __bf16* __restrict__ Wt,
    const float* __restrict__ bias, float* __restrict__ out,
    int mode, const float* __restrict__ xold, const float* __restrict__ skip)
{
    __shared__ __bf16 As[16][K + 16];          // A tile staged as bf16; stride mult of 8
    const int tid  = threadIdx.x;
    const int row0 = blockIdx.x * 16;

#pragma unroll
    for (int idx = tid; idx < 16 * K; idx += 256) {
        int r = idx / K, c = idx - r * K;
        As[r][c] = f2bf(A[(size_t)(row0 + r) * K + c]);
    }
    __syncthreads();

    const int wave = tid >> 5;                 // 0..7 -> N tile
    const int lane = tid & 31;
    const int m16  = lane & 15;
    const int half = lane >> 4;
    const int n    = wave * 16 + m16;          // this lane's output column

    v8f c = {0.f, 0.f, 0.f, 0.f, 0.f, 0.f, 0.f, 0.f};
#pragma unroll
    for (int kk = 0; kk < K; kk += 32) {
        // A 16x32 bf16 fragment: per lane = 16 contiguous k at kk + half*16 (matches B)
        v8bf alo = *(const v8bf*)&As[m16][kk + half * 16];
        v8bf ahi = *(const v8bf*)&As[m16][kk + half * 16 + 8];
        v16bf a = __builtin_shufflevector(alo, ahi,
                                          0, 1, 2, 3, 4, 5, 6, 7,
                                          8, 9, 10, 11, 12, 13, 14, 15);
        // B 32x16 bf16 fragment: lane covers k = kk+half*16 .. +15 at column n,
        // contiguous in the transposed weight layout.
        const v8bf* bp = (const v8bf*)(Wt + (size_t)n * K + kk + half * 16);
        v8bf blo = bp[0], bhi = bp[1];
        v16bf b = __builtin_shufflevector(blo, bhi,
                                          0, 1, 2, 3, 4, 5, 6, 7,
                                          8, 9, 10, 11, 12, 13, 14, 15);
        c = __builtin_amdgcn_wmma_f32_16x16x32_bf16(false, a, false, b,
                                                    (short)0, c, false, false);
    }

    float alpha = 0.f;
    if (mode == 2) alpha = 1.f / (1.f + __expf(-skip[0]));
    const float bn = bias[n];
#pragma unroll
    for (int j = 0; j < 8; ++j) {
        int m = half * 8 + j;                  // C/D layout: VGPR j -> row j / j+8
        size_t o = (size_t)(row0 + m) * 128 + n;
        float val = c[j] + bn;
        if (mode == 1)      val = val > 0.f ? val : 0.f;
        else if (mode == 2) val = alpha * val + (1.f - alpha) * xold[o];
        out[o] = val;
    }
}

// ---------------- per-node relation transform: out[n,h,e] = sum_d in[n,h,d]*R[h,d,e] ----------------
__global__ __launch_bounds__(256) void rel_transform(
    const float* __restrict__ in, const float* __restrict__ R,
    float* __restrict__ outp, int N)
{
    __shared__ float Rs[NHEAD * DH * DH];      // 8KB
    for (int i = threadIdx.x; i < NHEAD * DH * DH; i += 256) Rs[i] = R[i];
    __syncthreads();
    int gid = blockIdx.x * 256 + threadIdx.x;  // N*128 threads
    if (gid >= N * HID) return;
    int n = gid >> 7, he = gid & 127, h = he >> 4, e = he & 15;
    const float* xi = in + (size_t)n * HID + h * DH;
    const float* Rh = Rs + h * DH * DH + e;
    float s = 0.f;
#pragma unroll
    for (int d = 0; d < DH; ++d) s += xi[d] * Rh[d * DH];
    outp[gid] = s;
}

// ---------------- edge pass 1: logits + segment max ----------------
__global__ void edge_logits(const int* __restrict__ src, const int* __restrict__ dst,
                            const float* __restrict__ q, const float* __restrict__ krel,
                            const float* __restrict__ prel, float scale,
                            float* __restrict__ logits, unsigned* __restrict__ segmax)
{
    int gid = blockIdx.x * blockDim.x + threadIdx.x;   // NEDGE*8
    if (gid >= NEDGE * NHEAD) return;
    int e = gid >> 3, h = gid & 7;
    int s = src[e], d = dst[e];
    const float4* qp = (const float4*)(q    + (size_t)d * HID + h * DH);
    const float4* kp = (const float4*)(krel + (size_t)s * HID + h * DH);
    float acc = 0.f;
#pragma unroll
    for (int i = 0; i < 4; ++i) {
        float4 a = qp[i], b = kp[i];
        acc += a.x * b.x + a.y * b.y + a.z * b.z + a.w * b.w;
    }
    float lg = acc * prel[h] * scale;
    logits[gid] = lg;
    atomicMax(&segmax[(size_t)d * NHEAD + h], encf(lg));
}

// ---------------- edge pass 2: exp + segment sum ----------------
__global__ void edge_exp_sum(const int* __restrict__ dst,
                             float* __restrict__ logits,
                             const unsigned* __restrict__ segmax,
                             float* __restrict__ segsum)
{
    int gid = blockIdx.x * blockDim.x + threadIdx.x;
    if (gid >= NEDGE * NHEAD) return;
    int e = gid >> 3, h = gid & 7;
    int d = dst[e];
    float m  = decf(segmax[(size_t)d * NHEAD + h]);
    float ex = __expf(logits[gid] - m);
    logits[gid] = ex;                           // store numerator in place
    atomicAdd(&segsum[(size_t)d * NHEAD + h], ex);
}

// ---------------- edge pass 3: normalized message scatter ----------------
__global__ void edge_scatter(const int* __restrict__ src, const int* __restrict__ dst,
                             const float* __restrict__ logits,
                             const float* __restrict__ segsum,
                             const float* __restrict__ vrel, float* __restrict__ agg)
{
    int gid = blockIdx.x * blockDim.x + threadIdx.x;
    if (gid >= NEDGE * NHEAD) return;
    int e = gid >> 3, h = gid & 7;
    int s = src[e], d = dst[e];
    float w = logits[gid] / (segsum[(size_t)d * NHEAD + h] + 1e-16f);
    const float* vp = vrel + (size_t)s * HID + h * DH;
    float* ap = agg + (size_t)d * HID + h * DH;
#pragma unroll
    for (int i = 0; i < DH; ++i) atomicAdd(ap + i, w * vp[i]);
}

// ---------------- mean over edge types + exact GELU ----------------
__global__ void prep_gelu(const float* __restrict__ agg, float* __restrict__ g,
                          float invcnt, int n)
{
    int i = blockIdx.x * blockDim.x + threadIdx.x;
    if (i >= n) return;
    float x = agg[i] * invcnt;
    g[i] = 0.5f * x * (1.f + erff(x * 0.70710678118654752f));
}

// ---------------- host orchestration ----------------
static inline int cdiv(int a, int b) { return (a + b - 1) / b; }

extern "C" void kernel_launch(void* const* d_in, const int* in_sizes, int n_in,
                              void* d_out, int out_size, void* d_ws, size_t ws_size,
                              hipStream_t stream)
{
    (void)in_sizes; (void)n_in; (void)out_size; (void)ws_size;
    const float* x_paper  = (const float*)d_in[0];
    const float* x_author = (const float*)d_in[1];
    const float* Winp = (const float*)d_in[2];  const float* binp = (const float*)d_in[3];
    const float* Wina = (const float*)d_in[4];  const float* bina = (const float*)d_in[5];
    const float* Wk = (const float*)d_in[6];    const float* bk = (const float*)d_in[7];
    const float* Wq = (const float*)d_in[8];    const float* bq = (const float*)d_in[9];
    const float* Wv = (const float*)d_in[10];   const float* bv = (const float*)d_in[11];
    const float* Wa = (const float*)d_in[12];   const float* ba = (const float*)d_in[13];
    const float* skip  = (const float*)d_in[14];
    const float* a_rel = (const float*)d_in[15];
    const float* m_rel = (const float*)d_in[16];
    const float* p_rel = (const float*)d_in[17];
    const int* edge_pp = (const int*)d_in[18];
    const int* edge_ap = (const int*)d_in[19];
    const int* edge_pa = (const int*)d_in[20];
    float* out = (float*)d_out;

    const size_t NPF = (size_t)NP * HID, NAF = (size_t)NA * HID;

    // workspace carve-up (256B aligned)
    char* p = (char*)d_ws;
    auto alloc = [&](size_t bytes) { void* r = (void*)p; p += (bytes + 255) & ~(size_t)255; return r; };
    float* cur[2]  = { (float*)alloc(NPF * 4), (float*)alloc(NAF * 4) };
    float* kb[2]   = { (float*)alloc(NPF * 4), (float*)alloc(NAF * 4) };
    float* qb[2]   = { (float*)alloc(NPF * 4), (float*)alloc(NAF * 4) };
    float* vb[2]   = { (float*)alloc(NPF * 4), (float*)alloc(NAF * 4) };
    float* krel    = (float*)alloc(NPF * 4);
    float* vrel    = (float*)alloc(NPF * 4);
    float* gbuf[2] = { (float*)alloc(NPF * 4), (float*)alloc(NAF * 4) };
    float* agg[2]  = { (float*)alloc(NPF * 4), (float*)alloc(NAF * 4) };
    float* logits  = (float*)alloc((size_t)NEDGE * NHEAD * 4);
    unsigned* segmax = (unsigned*)alloc((size_t)NP * NHEAD * 4);
    float*    segsum = (float*)alloc((size_t)NP * NHEAD * 4);
    __bf16* wWinp = (__bf16*)alloc(IN_P * HID * 2);
    __bf16* wWina = (__bf16*)alloc(IN_A * HID * 2);
    __bf16* wWk   = (__bf16*)alloc(NLAYER * 2 * HID * HID * 2);
    __bf16* wWq   = (__bf16*)alloc(NLAYER * 2 * HID * HID * 2);
    __bf16* wWv   = (__bf16*)alloc(NLAYER * 2 * HID * HID * 2);
    __bf16* wWa   = (__bf16*)alloc(NLAYER * 2 * HID * HID * 2);

    const int Nn[2] = { NP, NA };

    // weight conversion f32 -> transposed bf16 ([128][K] per matrix), every call
    cvt_transpose_bf16<<<cdiv(IN_P * HID, 256), 256, 0, stream>>>(Winp, wWinp, IN_P, IN_P * HID);
    cvt_transpose_bf16<<<cdiv(IN_A * HID, 256), 256, 0, stream>>>(Wina, wWina, IN_A, IN_A * HID);
    const int WSZ = NLAYER * 2 * HID * HID;
    cvt_transpose_bf16<<<cdiv(WSZ, 256), 256, 0, stream>>>(Wk, wWk, HID, WSZ);
    cvt_transpose_bf16<<<cdiv(WSZ, 256), 256, 0, stream>>>(Wq, wWq, HID, WSZ);
    cvt_transpose_bf16<<<cdiv(WSZ, 256), 256, 0, stream>>>(Wv, wWv, HID, WSZ);
    cvt_transpose_bf16<<<cdiv(WSZ, 256), 256, 0, stream>>>(Wa, wWa, HID, WSZ);

    // input projection + relu
    gemm_bf16_n128<IN_P><<<NP / 16, 256, 0, stream>>>(x_paper,  wWinp, binp, cur[0], 1, nullptr, nullptr);
    gemm_bf16_n128<IN_A><<<NA / 16, 256, 0, stream>>>(x_author, wWina, bina, cur[1], 1, nullptr, nullptr);

    const int st_[3] = { 0, 1, 0 }, dt_[3] = { 0, 0, 1 };
    const int* eptr[3] = { edge_pp, edge_ap, edge_pa };
    const float scale = 0.25f;                 // 1/sqrt(D)
    const int EG = cdiv(NEDGE * NHEAD, 256);

    for (int l = 0; l < NLAYER; ++l) {
        // K / Q / V projections (WMMA)
        for (int t = 0; t < 2; ++t) {
            const int wo = (l * 2 + t) * HID * HID, bo = (l * 2 + t) * HID;
            gemm_bf16_n128<HID><<<Nn[t] / 16, 256, 0, stream>>>(cur[t], wWk + wo, bk + bo, kb[t], 0, nullptr, nullptr);
            gemm_bf16_n128<HID><<<Nn[t] / 16, 256, 0, stream>>>(cur[t], wWq + wo, bq + bo, qb[t], 0, nullptr, nullptr);
            gemm_bf16_n128<HID><<<Nn[t] / 16, 256, 0, stream>>>(cur[t], wWv + wo, bv + bo, vb[t], 0, nullptr, nullptr);
        }
        fill_f32<<<cdiv((int)NPF, 256), 256, 0, stream>>>(agg[0], 0.f, (int)NPF);
        fill_f32<<<cdiv((int)NAF, 256), 256, 0, stream>>>(agg[1], 0.f, (int)NAF);

        for (int r = 0; r < 3; ++r) {
            const int st = st_[r], dt = dt_[r];
            const int Ns = Nn[st], Nd = Nn[dt];
            const int* src = eptr[r];
            const int* dst = eptr[r] + NEDGE;
            const float* Ar = a_rel + (size_t)(l * 3 + r) * NHEAD * DH * DH;
            const float* Mr = m_rel + (size_t)(l * 3 + r) * NHEAD * DH * DH;
            const float* Pr = p_rel + (size_t)(l * 3 + r) * NHEAD;

            rel_transform<<<cdiv(Ns * HID, 256), 256, 0, stream>>>(kb[st], Ar, krel, Ns);
            rel_transform<<<cdiv(Ns * HID, 256), 256, 0, stream>>>(vb[st], Mr, vrel, Ns);
            fill_u32<<<cdiv(Nd * NHEAD, 256), 256, 0, stream>>>(segmax, 0u, Nd * NHEAD);
            fill_f32<<<cdiv(Nd * NHEAD, 256), 256, 0, stream>>>(segsum, 0.f, Nd * NHEAD);

            edge_logits <<<EG, 256, 0, stream>>>(src, dst, qb[dt], krel, Pr, scale, logits, segmax);
            edge_exp_sum<<<EG, 256, 0, stream>>>(dst, logits, segmax, segsum);
            edge_scatter<<<EG, 256, 0, stream>>>(src, dst, logits, segsum, vrel, agg[dt]);
        }

        // mean over edge types (paper receives 2, author receives 1) + exact GELU
        prep_gelu<<<cdiv((int)NPF, 256), 256, 0, stream>>>(agg[0], gbuf[0], 0.5f, (int)NPF);
        prep_gelu<<<cdiv((int)NAF, 256), 256, 0, stream>>>(agg[1], gbuf[1], 1.0f, (int)NAF);

        // A-linear with fused sigmoid-skip blend; last layer writes d_out directly
        for (int t = 0; t < 2; ++t) {
            const int wo = (l * 2 + t) * HID * HID, bo = (l * 2 + t) * HID;
            float* dstp = (l == NLAYER - 1) ? (t == 0 ? out : out + NPF) : cur[t];
            gemm_bf16_n128<HID><<<Nn[t] / 16, 256, 0, stream>>>(gbuf[t], wWa + wo, ba + bo, dstp,
                                                                2, cur[t], skip + l * 2 + t);
        }
    }
}
